// DecoderBlock_44092134261444
// MI455X (gfx1250) — matrix operand order repaired and checked by reading, not compiled, not run on hardware
//
#include <hip/hip_runtime.h>
#include <hip/hip_bf16.h>

typedef __attribute__((ext_vector_type(16))) __bf16 v16bf;
typedef __attribute__((ext_vector_type(8)))  float  v8f;
typedef unsigned int u32x4 __attribute__((ext_vector_type(4)));
typedef int i32x4 __attribute__((ext_vector_type(4)));
typedef int i32x8 __attribute__((ext_vector_type(8)));

#define Bn 8
#define Sn 1024
#define Dn 768
#define Hn 8
#define DKn 96
#define FFn 1024

#if defined(__has_builtin)
#if __has_builtin(__builtin_amdgcn_tensor_load_to_lds) && __has_builtin(__builtin_amdgcn_s_wait_tensorcnt)
#define HAVE_TDM 1
#endif
#endif

union Frag {
  v16bf v;
  uint4 q[2];
  __bf16 h[16];
};

__device__ inline __bf16 f2bf(float f) {
  union { float f; unsigned u; } c; c.f = f;
  unsigned r = c.u + 0x7FFFu + ((c.u >> 16) & 1u);   // round-to-nearest-even
  unsigned short hs = (unsigned short)(r >> 16);
  union { unsigned short u; __bf16 b; } o; o.u = hs;
  return o.b;
}

#ifdef HAVE_TDM
// TDM: DMA a 2D tile (tileK halves contiguous x tileN rows, row pitch K halves)
// from global into LDS at byte offset lds_off, inserting 16B of padding after
// every 64B K-row so the LDS column pitch is 80B (bank-conflict-free for
// ds_load_b128, still 16B aligned).  D# bit layout per CDNA5 ISA ch.8.
__device__ inline void tdm_load_tileB(const __bf16* gsrc, unsigned lds_off,
                                      int tileK, int tileN, int K, int N) {
  unsigned long long ga = (unsigned long long)(__UINTPTR_TYPE__)gsrc;
  u32x4 g0;
  g0[0] = 1u;                                      // count=1 (valid user D#)
  g0[1] = lds_off;                                 // lds_addr (shared block @ 0)
  g0[2] = (unsigned)(ga & 0xFFFFFFFFu);            // global_addr[31:0]
  g0[3] = (unsigned)((ga >> 32) & 0x1FFFFFFu) | (2u << 30);  // addr[56:32], type=2
  i32x8 g1;
  g1[0] = (1 << 16)        // data_size = 2 bytes
        | (1 << 20)        // pad_enable
        | (3 << 22)        // pad_interval: 16 DWORDs (one 64B K-row)
        | (3 << 25);       // pad_amount: 4 DWORDs = 16B  -> 80B column pitch
  g1[1] = (K & 0xFFFF) << 16;                            // tensor_dim0[15:0]
  g1[2] = ((K >> 16) & 0xFFFF) | ((N & 0xFFFF) << 16);   // dim0[31:16], dim1[15:0]
  g1[3] = ((N >> 16) & 0xFFFF) | ((tileK & 0xFFFF) << 16); // dim1[31:16], tile_dim0
  g1[4] = (tileN & 0xFFFF);                              // tile_dim1 (tile_dim2=0)
  g1[5] = K;                                             // tensor_dim0_stride[31:0]
  g1[6] = 0;                                             // stride0[47:32], dim1_stride lo
  g1[7] = 0;
  i32x4 z4 = {0, 0, 0, 0};
#if __clang_major__ >= 23
  i32x8 z8 = {0, 0, 0, 0, 0, 0, 0, 0};
  __builtin_amdgcn_tensor_load_to_lds(g0, g1, z4, z4, z8, 0);
#else
  __builtin_amdgcn_tensor_load_to_lds(g0, g1, z4, z4, 0);
#endif
}
#endif

// ---------------- casts ----------------
__global__ void cast_f32_bf16(const float* __restrict__ in, __bf16* __restrict__ out, size_t n) {
  size_t i = (size_t)blockIdx.x * blockDim.x + threadIdx.x;
  size_t st = (size_t)gridDim.x * blockDim.x;
  for (; i < n; i += st) out[i] = f2bf(in[i]);
}

// in: K x N row-major f32 ; out: N x K row-major bf16 (i.e. B^T)
__global__ void castT_f32_bf16(const float* __restrict__ in, __bf16* __restrict__ out, int K, int N) {
  size_t total = (size_t)K * N;
  size_t i = (size_t)blockIdx.x * blockDim.x + threadIdx.x;
  size_t st = (size_t)gridDim.x * blockDim.x;
  for (; i < total; i += st) {
    int n = (int)(i / K), k = (int)(i % K);
    out[(size_t)n * K + k] = f2bf(in[(size_t)k * N + n]);
  }
}

// ---------------- generic GEMM: C[M,N] = act(A[M,K] @ Bt[N,K]^T + bias) ----------------
// block = 256 threads = 8 waves; block tile 128x64, wave tile 16x64.
// B panel is shared by all 8 waves -> staged once per K-step through LDS by the
// Tensor Data Mover (double-buffered on TENSORcnt); A fragments are register-prefetched.
__global__ __launch_bounds__(256) void gemm_bf16(
    const __bf16* __restrict__ A, const __bf16* __restrict__ Bt,
    const float* __restrict__ bias, float* __restrict__ Cf,
    __bf16* __restrict__ Cb, int M, int N, int K, int relu) {
  __shared__ __bf16 lb[2][64 * 40];   // 64 cols x 32 K-halves, 80B pitch
  int tid = threadIdx.x;
  int lane = tid & 31, w = tid >> 5;
  int lr = lane & 15, hl = lane >> 4;
  int row0 = blockIdx.y * 128 + w * 16;
  int col0 = blockIdx.x * 64;
  const int steps = K >> 5;

#ifndef HAVE_TDM
  int scol = tid >> 2, spart = tid & 3;
  const __bf16* gB = Bt + (size_t)(col0 + scol) * K + spart * 8;
#endif

  // stage 0
#ifdef HAVE_TDM
  if (w == 0) tdm_load_tileB(Bt + (size_t)col0 * K, 0u, 32, 64, K, N);
#else
  *(uint4*)&lb[0][scol * 40 + spart * 8] = *(const uint4*)(gB);
#endif

  v8f acc[4] = {};
  const __bf16* Ar = A + (size_t)(row0 + lr) * K;
  Frag a;
  a.q[0] = *(const uint4*)(Ar + 8 * hl);
  a.q[1] = *(const uint4*)(Ar + 8 * hl + 16);

  for (int s = 0; s < steps; ++s) {
    int nbuf = (s + 1) & 1;
#ifdef HAVE_TDM
    if (w == 0) {
      if (s + 1 < steps) {
        tdm_load_tileB(Bt + (size_t)col0 * K + (s + 1) * 32,
                       (unsigned)(nbuf * 5120), 32, 64, K, N);
        __builtin_amdgcn_s_wait_tensorcnt(1);   // stage s complete, s+1 in flight
      } else {
        __builtin_amdgcn_s_wait_tensorcnt(0);
      }
    }
#else
    if (s + 1 < steps)
      *(uint4*)&lb[nbuf][scol * 40 + spart * 8] = *(const uint4*)(gB + (s + 1) * 32);
#endif
    __syncthreads();

    Frag an = a;                      // register-prefetch next A fragment
    if (s + 1 < steps) {
      an.q[0] = *(const uint4*)(Ar + (s + 1) * 32 + 8 * hl);
      an.q[1] = *(const uint4*)(Ar + (s + 1) * 32 + 8 * hl + 16);
    }
    const __bf16* lbase = &lb[s & 1][8 * hl];
#pragma unroll
    for (int j = 0; j < 4; ++j) {
      Frag b;
      const __bf16* p = lbase + (j * 16 + lr) * 40;
      b.q[0] = *(const uint4*)(p);
      b.q[1] = *(const uint4*)(p + 16);
      acc[j] = __builtin_amdgcn_wmma_f32_16x16x32_bf16(
          false, a.v, false, b.v, (short)0, acc[j], false, false);
    }
    a = an;
    __syncthreads();                  // buf[s&1] free for stage s+2
  }

#pragma unroll
  for (int j = 0; j < 4; ++j) {
    int n = col0 + j * 16 + lr;
    float bv = bias ? bias[n] : 0.f;
#pragma unroll
    for (int r = 0; r < 8; ++r) {
      int m = row0 + 8 * hl + r;
      float vv = acc[j][r] + bv;
      if (relu) vv = fmaxf(vv, 0.f);
      size_t idx = (size_t)m * N + n;
      if (Cf) Cf[idx] = vv;
      if (Cb) Cb[idx] = f2bf(vv);
    }
  }
}

// ---------------- attention scores: Sc[bh,q,k] = mask ? -1e9 : q.k/sqrt(DK) ----------------
__global__ __launch_bounds__(256) void attn_scores(
    const __bf16* __restrict__ Q, const __bf16* __restrict__ KV,
    const unsigned char* __restrict__ amask, float* __restrict__ Sc) {
  int lane = threadIdx.x & 31, w = threadIdx.x >> 5;
  int lr = lane & 15, hl = lane >> 4;
  int bh = blockIdx.z, b = bh >> 3, h = bh & 7;
  int q0 = (blockIdx.y * 8 + w) * 16;
  int k0 = blockIdx.x * 16;
  size_t base = (size_t)bh * Sn * Sn;

  if (k0 > q0 + 15) {  // fully causal-masked tile
#pragma unroll
    for (int r = 0; r < 8; ++r) {
      int m = q0 + 8 * hl + r;
      Sc[base + (size_t)m * Sn + k0 + lr] = -1.0e9f;
    }
    return;
  }
  v8f acc = {};
  const __bf16* Qr = Q  + (size_t)(b * Sn + q0 + lr) * Dn + h * DKn;
  const __bf16* Kr = KV + (size_t)(b * Sn + k0 + lr) * Dn + h * DKn;
#pragma unroll
  for (int kk = 0; kk < DKn; kk += 32) {
    Frag a, bb;
    a.q[0]  = *(const uint4*)(Qr + kk + 8 * hl);
    a.q[1]  = *(const uint4*)(Qr + kk + 8 * hl + 16);
    bb.q[0] = *(const uint4*)(Kr + kk + 8 * hl);
    bb.q[1] = *(const uint4*)(Kr + kk + 8 * hl + 16);
    acc = __builtin_amdgcn_wmma_f32_16x16x32_bf16(
        false, a.v, false, bb.v, (short)0, acc, false, false);
  }
  const float scale = 0.10206207262f;  // 1/sqrt(96)
#pragma unroll
  for (int r = 0; r < 8; ++r) {
    int m = q0 + 8 * hl + r;
    int n = k0 + lr;
    bool msk = (amask[b * Sn + m] != 0) || (n > m);
    Sc[base + (size_t)m * Sn + n] = msk ? -1.0e9f : acc[r] * scale;
  }
}

// ---------------- per-COLUMN softmax stats (reference softmaxes over q axis) ----------------
__global__ __launch_bounds__(256) void col_softmax_stats(
    const float* __restrict__ Sc, float* __restrict__ cmax, float* __restrict__ crz) {
  int k = blockIdx.x * 256 + threadIdx.x;
  int bh = blockIdx.y;
  size_t base = (size_t)bh * Sn * Sn + k;
  float m = -3.0e38f;
  for (int q = 0; q < Sn; ++q) m = fmaxf(m, Sc[base + (size_t)q * Sn]);
  float s = 0.f;
  for (int q = 0; q < Sn; ++q) s += __expf(Sc[base + (size_t)q * Sn] - m);
  cmax[bh * Sn + k] = m;
  crz[bh * Sn + k] = (s > 0.f) ? 1.f / s : 0.f;
}

// ---------------- O[q,d] = sum_k softmax_col(Sc)[q,k] * V[k,d] ----------------
__global__ __launch_bounds__(256) void attn_out(
    const float* __restrict__ Sc, const __bf16* __restrict__ V,
    const float* __restrict__ cmax, const float* __restrict__ crz,
    float* __restrict__ O) {
  __shared__ float lm[Sn];
  __shared__ float lz[Sn];
  int bh = blockIdx.z, b = bh >> 3, h = bh & 7;
  for (int i = threadIdx.x; i < Sn; i += 256) {
    lm[i] = cmax[bh * Sn + i];
    lz[i] = crz[bh * Sn + i];
  }
  __syncthreads();
  int lane = threadIdx.x & 31, w = threadIdx.x >> 5;
  int lr = lane & 15, hl = lane >> 4;
  int q0 = (blockIdx.y * 8 + w) * 16;
  int d0 = blockIdx.x * 16;
  size_t srow = (size_t)bh * Sn * Sn + (size_t)(q0 + lr) * Sn;

  v8f acc = {};
  for (int kk = 0; kk <= q0 + 15; kk += 32) {  // causal: columns > q0+15 contribute ~0
    Frag a, bb;
#pragma unroll
    for (int e = 0; e < 16; ++e) {
      int kidx = kk + ((e < 8) ? (8 * hl + e) : (16 + 8 * hl + (e - 8)));
      float sv = Sc[srow + kidx];
      a.h[e] = f2bf(__expf(sv - lm[kidx]) * lz[kidx]);
    }
#pragma unroll
    for (int e = 0; e < 16; ++e) {
      int kidx = kk + ((e < 8) ? (8 * hl + e) : (16 + 8 * hl + (e - 8)));
      bb.h[e] = V[(size_t)(b * Sn + kidx) * Dn + h * DKn + d0 + lr];
    }
    acc = __builtin_amdgcn_wmma_f32_16x16x32_bf16(
        false, a.v, false, bb.v, (short)0, acc, false, false);
  }
#pragma unroll
  for (int r = 0; r < 8; ++r) {
    int m = q0 + 8 * hl + r;
    O[(size_t)(b * Sn + m) * Dn + h * DKn + d0 + lr] = acc[r];
  }
}

// ---------------- y = LN(Xa + R)*g + be ; also bf16 copy for next GEMM ----------------
__global__ __launch_bounds__(256) void ln_residual(
    const float* __restrict__ Xa, const float* __restrict__ R,
    const float* __restrict__ g, const float* __restrict__ be,
    float* __restrict__ Y, __bf16* __restrict__ Yb) {
  __shared__ float s1[256];
  __shared__ float s2[256];
  int row = blockIdx.x, tid = threadIdx.x;
  size_t base = (size_t)row * Dn;
  float v[3], sum = 0.f, sq = 0.f;
#pragma unroll
  for (int i = 0; i < 3; ++i) {
    int c = tid + 256 * i;
    v[i] = Xa[base + c] + R[base + c];
    sum += v[i]; sq += v[i] * v[i];
  }
  s1[tid] = sum; s2[tid] = sq;
  __syncthreads();
  for (int off = 128; off > 0; off >>= 1) {
    if (tid < off) { s1[tid] += s1[tid + off]; s2[tid] += s2[tid + off]; }
    __syncthreads();
  }
  float mean = s1[0] * (1.f / Dn);
  float var  = s2[0] * (1.f / Dn) - mean * mean;
  float rs = rsqrtf(var + 1e-5f);
#pragma unroll
  for (int i = 0; i < 3; ++i) {
    int c = tid + 256 * i;
    float y = (v[i] - mean) * rs * g[c] + be[c];
    Y[base + c] = y;
    if (Yb) Yb[base + c] = f2bf(y);
  }
}

extern "C" void kernel_launch(void* const* d_in, const int* in_sizes, int n_in,
                              void* d_out, int out_size, void* d_ws, size_t ws_size,
                              hipStream_t stream) {
  (void)in_sizes; (void)n_in; (void)out_size; (void)ws_size;
  const float* x     = (const float*)d_in[0];
  const unsigned char* amask = (const unsigned char*)d_in[1];
  const float* a1_Wq = (const float*)d_in[2];
  const float* a1_bq = (const float*)d_in[3];
  const float* a1_Wv = (const float*)d_in[4];
  const float* a1_bv = (const float*)d_in[5];
  const float* a1_g  = (const float*)d_in[6];
  const float* a1_b  = (const float*)d_in[7];
  const float* a2_Wq = (const float*)d_in[8];
  const float* a2_bq = (const float*)d_in[9];
  const float* a2_Wv = (const float*)d_in[10];
  const float* a2_bv = (const float*)d_in[11];
  const float* a2_g  = (const float*)d_in[12];
  const float* a2_b  = (const float*)d_in[13];
  const float* f_W1  = (const float*)d_in[14];
  const float* f_b1  = (const float*)d_in[15];
  const float* f_W2  = (const float*)d_in[16];
  const float* f_b2  = (const float*)d_in[17];
  const float* f_g   = (const float*)d_in[18];
  const float* f_b   = (const float*)d_in[19];

  char* ws = (char*)d_ws;
  size_t off = 0;
  auto alloc = [&](size_t bytes) -> void* {
    void* p = ws + off;
    off += (bytes + 255) & ~(size_t)255;
    return p;
  };

  const size_t BSD = (size_t)Bn * Sn * Dn;
  __bf16* actX = (__bf16*)alloc(BSD * 2);
  __bf16* y1b  = (__bf16*)alloc(BSD * 2);
  __bf16* y2b  = (__bf16*)alloc(BSD * 2);
  float*  y1f  = (float*)alloc(BSD * 4);
  float*  y2f  = (float*)alloc(BSD * 4);
  __bf16* qb   = (__bf16*)alloc(BSD * 2);
  __bf16* kvb  = (__bf16*)alloc(BSD * 2);
  float*  sc   = (float*)alloc((size_t)Bn * Hn * Sn * Sn * 4);
  float*  cm   = (float*)alloc((size_t)Bn * Hn * Sn * 4);
  float*  cz   = (float*)alloc((size_t)Bn * Hn * Sn * 4);
  float*  obuf = (float*)alloc(BSD * 4);
  __bf16* hbuf = (__bf16*)alloc((size_t)Bn * Sn * FFn * 2);
  __bf16* wq1T = (__bf16*)alloc((size_t)Dn * Dn * 2);
  __bf16* wv1T = (__bf16*)alloc((size_t)Dn * Dn * 2);
  __bf16* wq2T = (__bf16*)alloc((size_t)Dn * Dn * 2);
  __bf16* wv2T = (__bf16*)alloc((size_t)Dn * Dn * 2);
  __bf16* w1T  = (__bf16*)alloc((size_t)Dn * FFn * 2);
  __bf16* w2T  = (__bf16*)alloc((size_t)FFn * Dn * 2);

  dim3 blk(256);
  cast_f32_bf16<<<2048, blk, 0, stream>>>(x, actX, BSD);
  castT_f32_bf16<<<2048, blk, 0, stream>>>(a1_Wq, wq1T, Dn, Dn);
  castT_f32_bf16<<<2048, blk, 0, stream>>>(a1_Wv, wv1T, Dn, Dn);
  castT_f32_bf16<<<2048, blk, 0, stream>>>(a2_Wq, wq2T, Dn, Dn);
  castT_f32_bf16<<<2048, blk, 0, stream>>>(a2_Wv, wv2T, Dn, Dn);
  castT_f32_bf16<<<2048, blk, 0, stream>>>(f_W1, w1T, Dn, FFn);
  castT_f32_bf16<<<2048, blk, 0, stream>>>(f_W2, w2T, FFn, Dn);

  const int M = Bn * Sn;
  dim3 gD(Dn / 64, M / 128);          // proj / ffn2 grid
  dim3 gF(FFn / 64, M / 128);         // ffn1 grid
  dim3 gsc(Sn / 16, Sn / 128, Bn * Hn);
  dim3 gcs(Sn / 256, Bn * Hn);
  dim3 gao(DKn / 16, Sn / 128, Bn * Hn);

  // ---- layer 1 MHA ----
  gemm_bf16<<<gD, blk, 0, stream>>>(actX, wq1T, a1_bq, nullptr, qb,  M, Dn, Dn, 0);
  gemm_bf16<<<gD, blk, 0, stream>>>(actX, wv1T, a1_bv, nullptr, kvb, M, Dn, Dn, 0);
  attn_scores<<<gsc, blk, 0, stream>>>(qb, kvb, amask, sc);
  col_softmax_stats<<<gcs, blk, 0, stream>>>(sc, cm, cz);
  attn_out<<<gao, blk, 0, stream>>>(sc, kvb, cm, cz, obuf);
  ln_residual<<<M, blk, 0, stream>>>(obuf, x, a1_g, a1_b, y1f, y1b);

  // ---- layer 2 MHA ----
  gemm_bf16<<<gD, blk, 0, stream>>>(y1b, wq2T, a2_bq, nullptr, qb,  M, Dn, Dn, 0);
  gemm_bf16<<<gD, blk, 0, stream>>>(y1b, wv2T, a2_bv, nullptr, kvb, M, Dn, Dn, 0);
  attn_scores<<<gsc, blk, 0, stream>>>(qb, kvb, amask, sc);
  col_softmax_stats<<<gcs, blk, 0, stream>>>(sc, cm, cz);
  attn_out<<<gao, blk, 0, stream>>>(sc, kvb, cm, cz, obuf);
  ln_residual<<<M, blk, 0, stream>>>(obuf, y1f, a2_g, a2_b, y2f, y2b);

  // ---- FFN ----
  gemm_bf16<<<gF, blk, 0, stream>>>(y2b, w1T, f_b1, nullptr, hbuf, M, FFn, Dn, 1);
  gemm_bf16<<<gD, blk, 0, stream>>>(hbuf, w2T, f_b2, obuf, nullptr, M, Dn, FFn, 0);
  ln_residual<<<M, blk, 0, stream>>>(obuf, y2f, f_g, f_b, (float*)d_out, nullptr);
}
